// ShapeGuidedOrientatedRWKV2D_15891378995284
// MI455X (gfx1250) — matrix operand order repaired and checked by reading, not compiled, not run on hardware
//
#include <hip/hip_runtime.h>
#include <hip/hip_bf16.h>

#define Bn 16
#define Cn 192
#define Hn 96
#define Wn 96
#define Ln (Hn*Wn)                 // 9216
#define NE ((long)Bn*(long)Ln*Cn)  // 28,311,552 elements per (dir-less) tensor
#define NCH 32                     // chunks per sequence for the parallel scan
#define CHn (Ln/NCH)               // 288 steps per chunk

typedef __attribute__((ext_vector_type(16))) __bf16 v16bf;
typedef __attribute__((ext_vector_type(8)))  float  v8f;

static __device__ __forceinline__ unsigned short f2bf(float f) {
  unsigned u = __builtin_bit_cast(unsigned, f);
  u += 0x7FFFu + ((u >> 16) & 1u);          // round-to-nearest-even
  return (unsigned short)(u >> 16);
}
static __device__ __forceinline__ __bf16 us2bf(unsigned short h) {
  return __builtin_bit_cast(__bf16, h);
}
static __device__ __forceinline__ float fastrcp(float x) {
  return __builtin_amdgcn_rcpf(x);          // v_rcp_f32: 1 trans op vs full IEEE div chain
}

// ---------------------------------------------------------------- saliency mean
__global__ __launch_bounds__(256) void sal_mean_kernel(const float* __restrict__ sal,
                                                       float* __restrict__ s) {
  int l = blockIdx.x * 256 + threadIdx.x;
  if (l >= Ln) return;
  float acc = 0.f;
  for (int b = 0; b < Bn; ++b) acc += sal[(long)b * Ln + l];
  s[l] = acc * (1.0f / Bn);
}

// ---------------------------------------------------------------- 4 scan orders (stable argsort via rank counting, saliency in LDS)
__global__ __launch_bounds__(256) void order_kernel(const float* __restrict__ s,
                                                    int* __restrict__ orders) {
  __shared__ float sh[Ln];
  for (int e = threadIdx.x; e < Ln; e += 256) sh[e] = s[e];
  __syncthreads();
  int j  = blockIdx.x * 256 + threadIdx.x;   // 0..Ln-1
  int oi = blockIdx.y;                        // 0..3
  bool colmajor = (oi == 1) || (oi == 3);
  bool desc     = (oi < 2);
  int jj = colmajor ? (j % Hn) * Wn + (j / Hn) : j;   // base_v mapping
  float key = sh[jj];
  int rank = 0;
  for (int j2 = 0; j2 < Ln; ++j2) {
    int jj2 = colmajor ? (j2 % Hn) * Wn + (j2 / Hn) : j2;
    float k2 = sh[jj2];
    bool before = desc ? ((k2 > key) || (k2 == key && j2 < j))
                       : ((k2 < key) || (k2 == key && j2 < j));
    rank += before ? 1 : 0;
  }
  orders[(long)oi * Ln + rank] = jj;
}

// ---------------------------------------------------------------- fp32 -> bf16 weight convert
__global__ __launch_bounds__(256) void f2bf_kernel(const float* __restrict__ src,
                                                   unsigned short* __restrict__ dst, int n) {
  int i = blockIdx.x * 256 + threadIdx.x;
  if (i < n) dst[i] = f2bf(src[i]);
}

// ---------------------------------------------------------------- depthwise conv mix -> bf16 (B,L,C) activation matrix
__global__ __launch_bounds__(256) void conv_pack_kernel(
    const float* __restrict__ x, const float* __restrict__ alpha,
    const float* __restrict__ w1, const float* __restrict__ w3,
    const float* __restrict__ w5, unsigned short* __restrict__ xseq) {
  long tid = (long)blockIdx.x * 256 + threadIdx.x;   // over B*C*L
  int l = (int)(tid % Ln);
  int c = (int)((tid / Ln) % Cn);
  int b = (int)(tid / ((long)Ln * Cn));
  int h = l / Wn, w = l % Wn;
  const float* xp = x + ((long)b * Cn + c) * Ln;
  float ctr = xp[l];
  float acc = (alpha[0] + alpha[1] * w1[c]) * ctr;
  float a2 = alpha[2];
  const float* k3 = w3 + c * 9;
  #pragma unroll
  for (int dy = -1; dy <= 1; ++dy)
    #pragma unroll
    for (int dx = -1; dx <= 1; ++dx) {
      int hh = h + dy, ww = w + dx;
      if (hh >= 0 && hh < Hn && ww >= 0 && ww < Wn)
        acc += a2 * k3[(dy + 1) * 3 + (dx + 1)] * xp[hh * Wn + ww];
    }
  float a4 = alpha[3];
  const float* k5 = w5 + c * 25;
  #pragma unroll
  for (int dy = -2; dy <= 2; ++dy)
    #pragma unroll
    for (int dx = -2; dx <= 2; ++dx) {
      int hh = h + dy, ww = w + dx;
      if (hh >= 0 && hh < Hn && ww >= 0 && ww < Wn)
        acc += a4 * k5[(dy + 2) * 5 + (dx + 2)] * xp[hh * Wn + ww];
    }
  xseq[((long)b * Ln + l) * Cn + c] = f2bf(acc);
}

// ---------------------------------------------------------------- async A-tile staging: memory -> LDS, ASYNCcnt-tracked
// Each of 384 threads moves one 16B slice of the contiguous 16x192 bf16 tile.
static __device__ __forceinline__ void stage_tile_async(unsigned short* As,
                                                        const unsigned short* gsrc) {
  unsigned long long gaddr =
      (unsigned long long)gsrc + (unsigned long long)threadIdx.x * 16ull;
  unsigned ldsoff = (unsigned)(unsigned long long)(&As[0]) + (unsigned)threadIdx.x * 16u;
  asm volatile("global_load_async_to_lds_b128 %0, %1, off"
               :: "v"(ldsoff), "v"(gaddr) : "memory");
  asm volatile("s_wait_asynccnt 0" ::: "memory");
  __syncthreads();
}

// ---------------------------------------------------------------- fused K/V/R projection GEMMs (bf16 WMMA, fp32 accum)
// grid: (M_tiles = B*L/16, 4 dirs); block: 384 = 12 waves, wave w owns N-tile w.
__global__ __launch_bounds__(384) void proj_wmma_kernel(
    const unsigned short* __restrict__ xseq,
    const unsigned short* __restrict__ wk16, const unsigned short* __restrict__ wv16,
    const unsigned short* __restrict__ wr16,
    float* __restrict__ kout, float* __restrict__ vout, float* __restrict__ srout) {
  __shared__ __align__(16) unsigned short As[16 * Cn];
  const int dir = blockIdx.y;
  const long mbase = (long)blockIdx.x * 16;
  const unsigned short* wkp = wk16 + (long)dir * Cn * Cn;
  const unsigned short* wvp = wv16 + (long)dir * Cn * Cn;
  const unsigned short* wrp = wr16 + (long)dir * Cn * Cn;
  float* kp  = kout  + (long)dir * NE;
  float* vp  = vout  + (long)dir * NE;
  float* srp = srout + (long)dir * NE;

  stage_tile_async(As, xseq + mbase * Cn);

  const int lane  = threadIdx.x & 31;
  const int wave  = threadIdx.x >> 5;
  const int mrow  = lane & 15;
  const int ahalf = (lane >> 4) * 8;   // unused: A also reads k = kb + bhalf + i, like B
  const int bhalf = (lane >> 4) * 16;  // B K-half offset
  const int ncol  = wave * 16 + (lane & 15);

  v8f ck = {}, cv = {}, cr = {};
  for (int kb = 0; kb < Cn; kb += 32) {
    v16bf a, bk, bv, br;
    #pragma unroll
    for (int i = 0; i < 16; ++i) {
      int kA = kb + bhalf + i;
      a[i] = us2bf(As[mrow * Cn + kA]);
      int kB = kb + bhalf + i;
      long wo = (long)ncol * Cn + kB;        // W[n][k] (x @ W^T)
      bk[i] = us2bf(wkp[wo]);
      bv[i] = us2bf(wvp[wo]);
      br[i] = us2bf(wrp[wo]);
    }
    ck = __builtin_amdgcn_wmma_f32_16x16x32_bf16(false, a, false, bk, (short)0, ck, false, false);
    cv = __builtin_amdgcn_wmma_f32_16x16x32_bf16(false, a, false, bv, (short)0, cv, false, false);
    cr = __builtin_amdgcn_wmma_f32_16x16x32_bf16(false, a, false, br, (short)0, cr, false, false);
  }
  const int moff = (lane >> 4) * 8;
  #pragma unroll
  for (int r = 0; r < 8; ++r) {
    long o = (mbase + r + moff) * Cn + ncol;
    kp[o]  = ck[r];
    vp[o]  = cv[r];
    srp[o] = fastrcp(1.0f + __expf(-cr[r]));   // sigmoid gate
  }
}

// ================================================================ WKV parallel chunked scan
// Log-compensated linear recurrence: semantic state Abar = aa*e^pp, Bbar = bb*e^pp with
//   Abar' = e^w * Abar + e^k * v ,  Bbar' = e^w * Bbar + e^k
// so per-chunk summaries (run from zero state) compose associatively across chunks.

// pass 1: per-chunk local summary (aa,bb,pp), zero-state start. grid (B, NCH, 4), block C.
__global__ __launch_bounds__(Cn) void wkv_chunk_kernel(
    const float* __restrict__ kbuf, const float* __restrict__ vbuf,
    const int* __restrict__ orders, const float* __restrict__ decay,
    float* __restrict__ sA, float* __restrict__ sB, float* __restrict__ sP) {
  const int c = threadIdx.x, b = blockIdx.x, m = blockIdx.y, dir = blockIdx.z;
  const int* ord = orders + (long)dir * Ln + m * CHn;
  const float* K = kbuf + (long)dir * NE;
  const float* V = vbuf + (long)dir * NE;
  const float w = decay[dir * Cn + c] * (1.0f / Ln);
  const long base = (long)b * Ln * Cn;
  float aa = 0.f, bb = 0.f, pp = -1e38f;
  for (int t = 0; t < CHn; ++t) {
    long o = base + (long)ord[t] * Cn + c;
    float kk = K[o], vv = V[o];
    float ww2 = w + pp;
    float p2  = fmaxf(ww2, kk);
    float f1 = __expf(ww2 - p2), f2 = __expf(kk - p2);
    aa = f1 * aa + f2 * vv;
    bb = f1 * bb + f2;
    pp = p2;
  }
  long so = (((long)dir * Bn + b) * NCH + m) * Cn + c;
  sA[so] = aa; sB[so] = bb; sP[so] = pp;
}

// pass 2: exclusive prefix-combine over the 32 chunk summaries (in place: slot m gets the
// state at the START of chunk m). One thread per (dir,b,c) lane.
__global__ __launch_bounds__(256) void wkv_prefix_kernel(
    float* __restrict__ sA, float* __restrict__ sB, float* __restrict__ sP,
    const float* __restrict__ decay) {
  int tid = blockIdx.x * 256 + threadIdx.x;       // over 4*B*C = 12288
  int c = tid % Cn, b = (tid / Cn) % Bn, dir = tid / (Cn * Bn);
  const float nw = (float)CHn * decay[dir * Cn + c] * (1.0f / Ln);  // chunk-level decay
  float A = 0.f, B = 0.f, P = -1e38f;
  for (int m = 0; m < NCH; ++m) {
    long so = (((long)dir * Bn + b) * NCH + m) * Cn + c;
    float ca = sA[so], cb = sB[so], q = sP[so];
    sA[so] = A; sB[so] = B; sP[so] = P;           // store chunk-entry state
    float Pw = P + nw;
    float Pn = fmaxf(Pw, q);
    float e0 = __expf(Pw - Pn), e1 = __expf(q - Pn);
    A = e0 * A + e1 * ca;
    B = e0 * B + e1 * cb;
    P = Pn;
  }
}

// pass 3: replay each chunk from its entry state, emit y in place over K.
__global__ __launch_bounds__(Cn) void wkv_apply_kernel(
    float* __restrict__ kbuf, const float* __restrict__ vbuf,
    const int* __restrict__ orders, const float* __restrict__ decay,
    const float* __restrict__ first,
    const float* __restrict__ sA, const float* __restrict__ sB,
    const float* __restrict__ sP) {
  const int c = threadIdx.x, b = blockIdx.x, m = blockIdx.y, dir = blockIdx.z;
  const int* ord = orders + (long)dir * Ln + m * CHn;
  float* K = kbuf + (long)dir * NE;
  const float* V = vbuf + (long)dir * NE;
  const float w = decay[dir * Cn + c] * (1.0f / Ln);
  const float u = first[dir * Cn + c] * (1.0f / Ln);
  const long base = (long)b * Ln * Cn;
  long so = (((long)dir * Bn + b) * NCH + m) * Cn + c;
  float aa = sA[so], bb = sB[so], pp = sP[so];
  for (int t = 0; t < CHn; ++t) {
    long o = base + (long)ord[t] * Cn + c;
    float kk = K[o], vv = V[o];
    float ww = u + kk;
    float p  = fmaxf(pp, ww);
    float e1 = __expf(pp - p), e2 = __expf(ww - p);
    float y  = (e1 * aa + e2 * vv) * fastrcp(e1 * bb + e2);
    float ww2 = w + pp;
    float p2  = fmaxf(ww2, kk);
    float f1 = __expf(ww2 - p2), f2 = __expf(kk - p2);
    aa = f1 * aa + f2 * vv;
    bb = f1 * bb + f2;
    pp = p2;
    K[o] = y;   // inverse-permute: write back at original position
  }
}

// ---------------------------------------------------------------- LayerNorm(C) + sigmoid gate -> bf16 A matrix
__global__ __launch_bounds__(256) void ln_gate_kernel(
    const float* __restrict__ ybuf, const float* __restrict__ srbuf,
    const float* __restrict__ g, const float* __restrict__ bta,
    unsigned short* __restrict__ abuf) {
  const int dir  = blockIdx.y;
  const long row = (long)blockIdx.x * 8 + (threadIdx.x >> 5);
  const int lane = threadIdx.x & 31;
  const float* yr  = ybuf  + (long)dir * NE + row * Cn;
  const float* srr = srbuf + (long)dir * NE + row * Cn;
  unsigned short* ar = abuf + (long)dir * NE + row * Cn;
  const float* gg = g   + dir * Cn;
  const float* bb = bta + dir * Cn;
  float v[6], s = 0.f, s2 = 0.f;
  #pragma unroll
  for (int t = 0; t < 6; ++t) {
    float f = yr[lane + 32 * t];
    v[t] = f; s += f; s2 += f * f;
  }
  #pragma unroll
  for (int off = 16; off; off >>= 1) {
    s  += __shfl_xor(s,  off, 32);
    s2 += __shfl_xor(s2, off, 32);
  }
  float mu  = s * (1.0f / Cn);
  float var = s2 * (1.0f / Cn) - mu * mu;
  float inv = rsqrtf(var + 1e-5f);
  #pragma unroll
  for (int t = 0; t < 6; ++t) {
    int c = lane + 32 * t;
    float yn = (v[t] - mu) * inv * gg[c] + bb[c];
    ar[c] = f2bf(srr[c] * yn);
  }
}

// ---------------------------------------------------------------- output GEMM (sr*ynorm) @ Wo^T -> grids
__global__ __launch_bounds__(384) void out_wmma_kernel(
    const unsigned short* __restrict__ abuf, const unsigned short* __restrict__ wo16,
    float* __restrict__ grids) {
  __shared__ __align__(16) unsigned short As[16 * Cn];
  const int dir = blockIdx.y;
  const long mbase = (long)blockIdx.x * 16;
  const unsigned short* wp = wo16 + (long)dir * Cn * Cn;
  float* gp = grids + (long)dir * NE;

  stage_tile_async(As, abuf + (long)dir * NE + mbase * Cn);

  const int lane  = threadIdx.x & 31;
  const int wave  = threadIdx.x >> 5;
  const int mrow  = lane & 15;
  const int ahalf = (lane >> 4) * 8;
  const int bhalf = (lane >> 4) * 16;
  const int ncol  = wave * 16 + (lane & 15);

  v8f acc = {};
  for (int kb = 0; kb < Cn; kb += 32) {
    v16bf a, bw;
    #pragma unroll
    for (int i = 0; i < 16; ++i) {
      int kA = kb + bhalf + i;
      a[i] = us2bf(As[mrow * Cn + kA]);
      int kB = kb + bhalf + i;
      bw[i] = us2bf(wp[(long)ncol * Cn + kB]);
    }
    acc = __builtin_amdgcn_wmma_f32_16x16x32_bf16(false, a, false, bw, (short)0, acc, false, false);
  }
  const int moff = (lane >> 4) * 8;
  #pragma unroll
  for (int r = 0; r < 8; ++r)
    gp[(mbase + r + moff) * Cn + ncol] = acc[r];
}

// ---------------------------------------------------------------- per-row channel mean of each grid
__global__ __launch_bounds__(256) void mean_kernel(const float* __restrict__ grids,
                                                   float* __restrict__ means) {
  const int dir  = blockIdx.y;
  const long row = (long)blockIdx.x * 8 + (threadIdx.x >> 5);
  const int lane = threadIdx.x & 31;
  const float* g = grids + (long)dir * NE + row * Cn;
  float s = 0.f;
  #pragma unroll
  for (int t = 0; t < 6; ++t) s += g[lane + 32 * t];
  #pragma unroll
  for (int off = 16; off; off >>= 1) s += __shfl_xor(s, off, 32);
  if (lane == 0) {
    int b = (int)(row / Ln), l = (int)(row % Ln);
    means[((long)b * 4 + dir) * Ln + l] = s * (1.0f / Cn);
  }
}

// ---------------------------------------------------------------- 4x4 mixing logits + softmax weights
__global__ __launch_bounds__(256) void mixw_kernel(const float* __restrict__ means,
                                                   const float* __restrict__ Wm,
                                                   const float* __restrict__ bm,
                                                   float* __restrict__ wts) {
  int tid = blockIdx.x * 256 + threadIdx.x;
  if (tid >= Bn * Ln) return;
  int b = tid / Ln, l = tid % Ln;
  float m[4];
  #pragma unroll
  for (int i = 0; i < 4; ++i) m[i] = means[((long)b * 4 + i) * Ln + l];
  float lg[4], mx = -3.4e38f;
  #pragma unroll
  for (int o = 0; o < 4; ++o) {
    float v = bm[o];
    #pragma unroll
    for (int i = 0; i < 4; ++i) v += Wm[o * 4 + i] * m[i];
    lg[o] = v;
    mx = fmaxf(mx, v);
  }
  float e[4], sum = 0.f;
  #pragma unroll
  for (int o = 0; o < 4; ++o) { e[o] = __expf(lg[o] - mx); sum += e[o]; }
  float inv = fastrcp(sum);
  #pragma unroll
  for (int o = 0; o < 4; ++o) wts[((long)b * 4 + o) * Ln + l] = e[o] * inv;
}

// ---------------------------------------------------------------- weighted grid combine -> (B,C,H,W)
__global__ __launch_bounds__(256) void mix_kernel(const float* __restrict__ grids,
                                                  const float* __restrict__ wts,
                                                  float* __restrict__ out) {
  long tid = (long)blockIdx.x * 256 + threadIdx.x;   // == (b*C + c)*L + l
  int l = (int)(tid % Ln);
  long bc = tid / Ln;
  int c = (int)(bc % Cn);
  int b = (int)(bc / Cn);
  long rowoff = ((long)b * Ln + l) * Cn + c;
  float acc = 0.f;
  #pragma unroll
  for (int i = 0; i < 4; ++i)
    acc += grids[(long)i * NE + rowoff] * wts[((long)b * 4 + i) * Ln + l];
  out[tid] = acc;
}

// ================================================================ launch
extern "C" void kernel_launch(void* const* d_in, const int* in_sizes, int n_in,
                              void* d_out, int out_size, void* d_ws, size_t ws_size,
                              hipStream_t stream) {
  const float* x     = (const float*)d_in[0];
  const float* sal   = (const float*)d_in[1];
  const float* alpha = (const float*)d_in[2];
  const float* w1    = (const float*)d_in[3];
  const float* w3    = (const float*)d_in[4];
  const float* w5    = (const float*)d_in[5];
  const float* decay = (const float*)d_in[6];
  const float* first = (const float*)d_in[7];
  const float* Wk    = (const float*)d_in[8];
  const float* Wv    = (const float*)d_in[9];
  const float* Wr    = (const float*)d_in[10];
  const float* Wo    = (const float*)d_in[11];
  const float* lng   = (const float*)d_in[12];
  const float* lnb   = (const float*)d_in[13];
  const float* Wm    = (const float*)d_in[14];
  const float* bm    = (const float*)d_in[15];
  float* out = (float*)d_out;

  char* p = (char*)d_ws;
  auto alloc = [&](size_t bytes) -> char* {
    char* r = p;
    p += (bytes + 255) & ~(size_t)255;
    return r;
  };
  float*          s_buf  = (float*)alloc((size_t)Ln * 4);
  int*            orders = (int*)alloc((size_t)4 * Ln * 4);
  unsigned short* xseq   = (unsigned short*)alloc((size_t)NE * 2);
  unsigned short* wk16   = (unsigned short*)alloc((size_t)4 * Cn * Cn * 2);
  unsigned short* wv16   = (unsigned short*)alloc((size_t)4 * Cn * Cn * 2);
  unsigned short* wr16   = (unsigned short*)alloc((size_t)4 * Cn * Cn * 2);
  unsigned short* wo16   = (unsigned short*)alloc((size_t)4 * Cn * Cn * 2);
  float*          Kb     = (float*)alloc((size_t)4 * NE * 4);   // reused as Y after scan
  float*          Vb     = (float*)alloc((size_t)4 * NE * 4);
  float*          SRb    = (float*)alloc((size_t)4 * NE * 4);
  unsigned short* Ab     = (unsigned short*)alloc((size_t)4 * NE * 2);
  float*          grids  = (float*)alloc((size_t)4 * NE * 4);
  float*          means  = (float*)alloc((size_t)Bn * 4 * Ln * 4);
  float*          wts    = (float*)alloc((size_t)Bn * 4 * Ln * 4);
  size_t nstate = (size_t)4 * Bn * NCH * Cn;        // 393,216 per plane
  float*          sA     = (float*)alloc(nstate * 4);
  float*          sB     = (float*)alloc(nstate * 4);
  float*          sP     = (float*)alloc(nstate * 4);

  // 1) saliency mean + scan orders
  sal_mean_kernel<<<(Ln + 255) / 256, 256, 0, stream>>>(sal, s_buf);
  order_kernel<<<dim3(Ln / 256, 4), 256, 0, stream>>>(s_buf, orders);

  // 2) bf16 weight staging
  const int nW = 4 * Cn * Cn;
  f2bf_kernel<<<(nW + 255) / 256, 256, 0, stream>>>(Wk, wk16, nW);
  f2bf_kernel<<<(nW + 255) / 256, 256, 0, stream>>>(Wv, wv16, nW);
  f2bf_kernel<<<(nW + 255) / 256, 256, 0, stream>>>(Wr, wr16, nW);
  f2bf_kernel<<<(nW + 255) / 256, 256, 0, stream>>>(Wo, wo16, nW);

  // 3) depthwise conv mix -> bf16 (B,L,C)
  conv_pack_kernel<<<(int)(NE / 256), 256, 0, stream>>>(x, alpha, w1, w3, w5, xseq);

  // 4) K/V/R projections for all 4 directions (bf16 WMMA)
  proj_wmma_kernel<<<dim3(Bn * Ln / 16, 4), 384, 0, stream>>>(xseq, wk16, wv16, wr16,
                                                              Kb, Vb, SRb);

  // 5) WKV: chunk summaries -> prefix combine -> replay with prefix states
  wkv_chunk_kernel<<<dim3(Bn, NCH, 4), Cn, 0, stream>>>(Kb, Vb, orders, decay, sA, sB, sP);
  wkv_prefix_kernel<<<(4 * Bn * Cn) / 256, 256, 0, stream>>>(sA, sB, sP, decay);
  wkv_apply_kernel<<<dim3(Bn, NCH, 4), Cn, 0, stream>>>(Kb, Vb, orders, decay, first,
                                                        sA, sB, sP);

  // 6) LayerNorm + gate -> bf16 A
  ln_gate_kernel<<<dim3(Bn * Ln / 8, 4), 256, 0, stream>>>(Kb, SRb, lng, lnb, Ab);

  // 7) output GEMMs (bf16 WMMA) -> grids
  out_wmma_kernel<<<dim3(Bn * Ln / 16, 4), 384, 0, stream>>>(Ab, wo16, grids);

  // 8) channel means, softmax mixing weights, weighted combine
  mean_kernel<<<dim3(Bn * Ln / 8, 4), 256, 0, stream>>>(grids, means);
  mixw_kernel<<<(Bn * Ln + 255) / 256, 256, 0, stream>>>(means, Wm, bm, wts);
  mix_kernel<<<(int)(NE / 256), 256, 0, stream>>>(grids, wts, out);
}
